// GatingMixedDecoder_26242250179135
// MI455X (gfx1250) — compile-verified, matrix operand order checked by reading, not run on hardware
//
#include <hip/hip_runtime.h>

typedef __attribute__((ext_vector_type(16))) _Float16 v16h;
typedef __attribute__((ext_vector_type(8)))  _Float16 v8h;
typedef __attribute__((ext_vector_type(2)))  _Float16 v2h;
typedef __attribute__((ext_vector_type(8)))  float    v8f;

// ---------------- geometry ----------------
#define TB          64          // rows per block
#define NWAVE       8           // 256 threads, wave32
#define RAW_STRIDE  160         // f32 raw concat(z, act) row stride
#define LN_STRIDE   168         // f16 layernormed row stride (16B aligned rows)
#define G_STRIDE    136         // f16 gate activation stride (16B aligned rows)

// fragment buffer offsets (f16 elements) inside d_ws
#define OFF_W0  0u          // 8*4*8*512 = 131072
#define OFF_W1  131072u     // 8*5*8*512 = 163840
#define OFF_W2  294912u
#define OFF_W3  458752u     // 8*5*4*512 = 81920
#define OFF_G0  540672u     // 4*8*512
#define OFF_G1  557056u
#define OFF_G2  573440u     // 4*1*512
#define NFRAGS  1124        // total 32-lane fragments

__device__ inline float eluf(float x) { return x > 0.f ? x : __expf(x) - 1.f; }

// Issue one global_prefetch_b8 per 128B line of [p, p+bytes), spread over 256 threads.
__device__ inline void prefetch_range(const void* p, unsigned bytes, int tid)
{
  const char* cp = (const char*)p;
  for (unsigned off = (unsigned)tid * 128u; off < bytes; off += 256u * 128u)
    __builtin_prefetch(cp + off, 0, 3);
}

// =====================================================================
// Prep kernel: fp32 weights -> f16 WMMA B-matrix fragments.
// Fragment = [32 lanes][16 f16]; lane holds column colb+lane%16 with
// J-indices jb..jb+7 then jb+16..jb+23 (jb = q*32 + 8*(lane>=16)),
// matching the CDNA5 16-bit B 32x16 VGPR layout.
// =====================================================================
__global__ __launch_bounds__(256) void prep_frags(
    const float* __restrict__ w0, const float* __restrict__ w1,
    const float* __restrict__ w2, const float* __restrict__ w3,
    const float* __restrict__ gw0, const float* __restrict__ gw1,
    const float* __restrict__ gw2, _Float16* __restrict__ frag)
{
  int t = blockIdx.x * 256 + threadIdx.x;
  int fi = t >> 5, lane = t & 31;
  if (fi >= NFRAGS) return;

  const float* wp; int J, Kv, kch, ct; unsigned base; int f;
  if      (fi < 256)  { wp = w0;  J = 128; Kv = 128; kch = 4; ct = 8; base = OFF_W0; f = fi; }
  else if (fi < 576)  { wp = w1;  J = 160; Kv = 128; kch = 5; ct = 8; base = OFF_W1; f = fi - 256; }
  else if (fi < 896)  { wp = w2;  J = 160; Kv = 128; kch = 5; ct = 8; base = OFF_W2; f = fi - 576; }
  else if (fi < 1056) { wp = w3;  J = 160; Kv = 64;  kch = 5; ct = 4; base = OFF_W3; f = fi - 896; }
  else if (fi < 1088) { wp = gw0; J = 128; Kv = 128; kch = 4; ct = 8; base = OFF_G0; f = fi - 1056; }
  else if (fi < 1120) { wp = gw1; J = 128; Kv = 128; kch = 4; ct = 8; base = OFF_G1; f = fi - 1088; }
  else                { wp = gw2; J = 128; Kv = 8;   kch = 4; ct = 1; base = OFF_G2; f = fi - 1120; }

  int e = f / (kch * ct);
  int rem = f % (kch * ct);
  int q = rem / ct, ctile = rem % ct;
  int lane16 = lane & 15, hi = lane >> 4;
  int col = ctile * 16 + lane16;
  int jb  = q * 32 + hi * 8;

  v16h o;
#pragma unroll
  for (int i = 0; i < 16; ++i) {
    int j = jb + (i < 8 ? i : 8 + i);      // K offsets 0..7 then 16..23
    float v = 0.f;
    if (col < Kv) v = wp[((size_t)e * J + j) * Kv + col];
    o[i] = (_Float16)v;
  }
  *(v16h*)(frag + (size_t)base + (size_t)f * 512 + lane * 16) = o;
}

// ---------------- A-fragment load from LDS (row-major f16) ----------------
__device__ inline v16h load_afrag(const _Float16* sIn, int stride,
                                  int rowb, int q, int lane16, int hi)
{
  int j0 = q * 32 + hi * 8;
  const _Float16* ap = sIn + (rowb + lane16) * stride + j0;
  v8h lo = *(const v8h*)ap;          // K j0..j0+7   -> vgpr 0..3
  v8h hh = *(const v8h*)(ap + 16);   // K j0+16..+23 -> vgpr 4..7
  return __builtin_shufflevector(lo, hh, 0,1,2,3,4,5,6,7,8,9,10,11,12,13,14,15);
}

// ---------------- gate GEMM layer (ELU) ----------------
__device__ inline void gate_gemm(const _Float16* sIn, _Float16* sOut,
                                 const _Float16* frag, const float* gbias,
                                 int kch, int ctiles, int wave, int lane)
{
  int lane16 = lane & 15, hi = lane >> 4;
  const v16h* fp = (const v16h*)frag;
  int ntiles = 4 * ctiles;
  for (int t = wave; t < ntiles; t += NWAVE) {
    int rt = t / ctiles, ctile = t % ctiles;
    int rowb = rt * 16, colb = ctile * 16;
    v8f acc = {};
    for (int q = 0; q < kch; ++q) {
      v16h a = load_afrag(sIn, G_STRIDE, rowb, q, lane16, hi);
      v16h b = fp[(q * ctiles + ctile) * 32 + lane];
      acc = __builtin_amdgcn_wmma_f32_16x16x32_f16(false, a, false, b,
                                                   (short)0, acc, false, false);
    }
    float bn = gbias[colb + lane16];
#pragma unroll
    for (int r = 0; r < 8; ++r) {
      float v = eluf(acc[r] + bn);
      sOut[(rowb + r + hi * 8) * G_STRIDE + colb + lane16] = (_Float16)v;
    }
  }
}

// ---------------- soft-MoE expert layer ----------------
// out[b,k] = sum_e coeff[b,e] * ( LN(x)_b @ W_e + bias_e )[k]
// A-fragments loaded once per tile and reused across all 8 experts;
// per-wave column tile is loop-invariant so the compiler hoists all
// B-fragments into registers and reuses them across the 4 row tiles.
__device__ inline void expert_layer(const _Float16* sLN, const _Float16* frag,
                                    const float* ebias, const float* sCoeffT,
                                    float* sRaw, float* gout,
                                    int K, int kch, int ctiles,
                                    bool act, bool last, int rowBase,
                                    int wave, int lane)
{
  int lane16 = lane & 15, hi = lane >> 4;
  const v16h* fp = (const v16h*)frag;
  int ntiles = 4 * ctiles;
  for (int t = wave; t < ntiles; t += NWAVE) {
    int rt = t / ctiles, ctile = t % ctiles;
    int rowb = rt * 16, colb = ctile * 16;

    v16h av[5];                                  // A-frags reused by all 8 experts
    for (int q = 0; q < kch; ++q)
      av[q] = load_afrag(sLN, LN_STRIDE, rowb, q, lane16, hi);

    v8f acc = {};
    for (int e = 0; e < 8; ++e) {
      v8f tmp = {};
      for (int q = 0; q < kch; ++q) {
        v16h b = fp[((e * kch + q) * ctiles + ctile) * 32 + lane];
        tmp = __builtin_amdgcn_wmma_f32_16x16x32_f16(false, av[q], false, b,
                                                     (short)0, tmp, false, false);
      }
      float bn = ebias[e * K + colb + lane16];
#pragma unroll
      for (int r = 0; r < 8; ++r) {
        float cf = sCoeffT[e * TB + rowb + r + hi * 8];   // broadcast read
        acc[r] += cf * (tmp[r] + bn);
      }
    }
#pragma unroll
    for (int r = 0; r < 8; ++r) {
      int row = rowb + r + hi * 8;
      float v = acc[r];
      if (!last)
        sRaw[row * RAW_STRIDE + 32 + colb + lane16] = act ? eluf(v) : v;
      else
        gout[(size_t)(rowBase + row) * 64 + colb + lane16] = v;
    }
  }
}

// ---------------- per-row layernorm: sRaw(f32) -> sLN(f16) ----------------
// Vectorized: float4 reads, packed 2xf16 writes. J is 128 or 160 (mult of 4).
__device__ inline void ln_pass(const float* sRaw, _Float16* sLN, int J, int tid)
{
  if (tid < TB) {
    const float4* rp4 = (const float4*)(sRaw + tid * RAW_STRIDE);
    int n4 = J >> 2;
    float s = 0.f, ss = 0.f;
    for (int j = 0; j < n4; ++j) {
      float4 v = rp4[j];
      s  += v.x + v.y + v.z + v.w;
      ss += v.x * v.x + v.y * v.y + v.z * v.z + v.w * v.w;
    }
    float mean = s * (1.f / J);
    float var  = ss * (1.f / J) - mean * mean;
    float rstd = rsqrtf(var + 1e-5f);
    _Float16* op = sLN + tid * LN_STRIDE;
    for (int j = 0; j < n4; ++j) {
      float4 v = rp4[j];
      v2h p0, p1;
      p0[0] = (_Float16)((v.x - mean) * rstd);
      p0[1] = (_Float16)((v.y - mean) * rstd);
      p1[0] = (_Float16)((v.z - mean) * rstd);
      p1[1] = (_Float16)((v.w - mean) * rstd);
      *(v2h*)(op + j * 4)     = p0;
      *(v2h*)(op + j * 4 + 2) = p1;
    }
  }
}

// =====================================================================
// Main fused kernel: gate MLP + softmax + 4 soft-MoE expert layers
// =====================================================================
__global__ __launch_bounds__(256) void moe_main(
    const float* __restrict__ z,  const float* __restrict__ c,
    const float* __restrict__ b0, const float* __restrict__ b1,
    const float* __restrict__ b2, const float* __restrict__ b3,
    const float* __restrict__ gb0, const float* __restrict__ gb1,
    const float* __restrict__ gb2,
    const _Float16* __restrict__ frag, float* __restrict__ out)
{
  __shared__ float    sRaw[TB * RAW_STRIDE];   // 40960 B  raw concat(z, act)
  __shared__ _Float16 sLN [TB * LN_STRIDE];    // 21504 B  layernormed f16
  __shared__ _Float16 sG0 [TB * G_STRIDE];     // 17408 B  gate ping
  __shared__ _Float16 sG1 [TB * G_STRIDE];     // 17408 B  gate pong
  __shared__ float    sCoeffT[8 * TB];         //  2048 B  coeff^T (logits)

  int tid = threadIdx.x;
  int wave = tid >> 5, lane = tid & 31;
  int lane16 = lane & 15, hi = lane >> 4;
  int rowBase = blockIdx.x * TB;

  // ---- load z (cols 0..31) and c (cols 32..127) ----
  for (int i = tid; i < TB * 32; i += 256) {
    int r = i >> 5, j = i & 31;
    sRaw[r * RAW_STRIDE + j] = z[(size_t)(rowBase + r) * 32 + j];
  }
  for (int i = tid; i < TB * 96; i += 256) {
    int r = i / 96, j = i % 96;
    sRaw[r * RAW_STRIDE + 32 + j] = c[(size_t)(rowBase + r) * 96 + j];
  }
  // warm L2/WGP$ with layer-0 expert fragments while the gate runs
  prefetch_range(frag + OFF_W0, (OFF_W1 - OFF_W0) * 2u, tid);
  __syncthreads();

  // ---- x -> f16 for gate, and LN(x) (J=128) for expert layer 0 ----
  for (int i = tid; i < TB * 128; i += 256) {
    int r = i >> 7, j = i & 127;
    sG0[r * G_STRIDE + j] = (_Float16)sRaw[r * RAW_STRIDE + j];
  }
  ln_pass(sRaw, sLN, 128, tid);
  __syncthreads();

  // ---- gate MLP layers 0,1 ----
  gate_gemm(sG0, sG1, frag + OFF_G0, gb0, 4, 8, wave, lane);
  __syncthreads();
  gate_gemm(sG1, sG0, frag + OFF_G1, gb1, 4, 8, wave, lane);
  __syncthreads();

  // ---- gate layer 2 -> logits into sCoeffT (transposed [E][row]) ----
  {
    const v16h* fp = (const v16h*)(frag + OFF_G2);
    for (int t = wave; t < 4; t += NWAVE) {
      int rowb = t * 16;
      v8f acc = {};
      for (int q = 0; q < 4; ++q) {
        v16h a = load_afrag(sG0, G_STRIDE, rowb, q, lane16, hi);
        v16h b = fp[q * 32 + lane];
        acc = __builtin_amdgcn_wmma_f32_16x16x32_f16(false, a, false, b,
                                                     (short)0, acc, false, false);
      }
      if (lane16 < 8) {
        float bn = gb2[lane16];
#pragma unroll
        for (int r = 0; r < 8; ++r)
          sCoeffT[lane16 * TB + rowb + r + hi * 8] = acc[r] + bn;
      }
    }
  }
  __syncthreads();

  // ---- softmax over E=8 per row ----
  if (tid < TB) {
    float l[8];
#pragma unroll
    for (int e = 0; e < 8; ++e) l[e] = sCoeffT[e * TB + tid];
    float m = l[0];
#pragma unroll
    for (int e = 1; e < 8; ++e) m = fmaxf(m, l[e]);
    float s = 0.f;
#pragma unroll
    for (int e = 0; e < 8; ++e) { l[e] = __expf(l[e] - m); s += l[e]; }
    float inv = 1.f / s;
#pragma unroll
    for (int e = 0; e < 8; ++e) sCoeffT[e * TB + tid] = l[e] * inv;
  }
  __syncthreads();

  // ---- expert layers (prefetch next layer's fragments during each phase) ----
  prefetch_range(frag + OFF_W1, (OFF_W2 - OFF_W1) * 2u, tid);
  expert_layer(sLN, frag + OFF_W0, b0, sCoeffT, sRaw, out, 128, 4, 8, true,  false, rowBase, wave, lane);
  __syncthreads();
  ln_pass(sRaw, sLN, 160, tid);
  __syncthreads();

  prefetch_range(frag + OFF_W2, (OFF_W3 - OFF_W2) * 2u, tid);
  expert_layer(sLN, frag + OFF_W1, b1, sCoeffT, sRaw, out, 128, 5, 8, true,  false, rowBase, wave, lane);
  __syncthreads();
  ln_pass(sRaw, sLN, 160, tid);
  __syncthreads();

  prefetch_range(frag + OFF_W3, (OFF_G0 - OFF_W3) * 2u, tid);
  expert_layer(sLN, frag + OFF_W2, b2, sCoeffT, sRaw, out, 128, 5, 8, true,  false, rowBase, wave, lane);
  __syncthreads();
  ln_pass(sRaw, sLN, 160, tid);
  __syncthreads();

  expert_layer(sLN, frag + OFF_W3, b3, sCoeffT, sRaw, out, 64,  5, 4, false, true,  rowBase, wave, lane);
}

// =====================================================================
extern "C" void kernel_launch(void* const* d_in, const int* in_sizes, int n_in,
                              void* d_out, int out_size, void* d_ws, size_t ws_size,
                              hipStream_t stream)
{
  const float* z   = (const float*)d_in[0];
  const float* c   = (const float*)d_in[1];
  const float* w0  = (const float*)d_in[2];
  const float* b0  = (const float*)d_in[3];
  const float* w1  = (const float*)d_in[4];
  const float* b1  = (const float*)d_in[5];
  const float* w2  = (const float*)d_in[6];
  const float* b2  = (const float*)d_in[7];
  const float* w3  = (const float*)d_in[8];
  const float* b3  = (const float*)d_in[9];
  const float* gw0 = (const float*)d_in[10];
  const float* gb0 = (const float*)d_in[11];
  const float* gw1 = (const float*)d_in[12];
  const float* gb1 = (const float*)d_in[13];
  const float* gw2 = (const float*)d_in[14];
  const float* gb2 = (const float*)d_in[15];

  _Float16* frag = (_Float16*)d_ws;   // needs ~1.10 MB scratch

  int prepBlocks = (NFRAGS * 32 + 255) / 256;
  prep_frags<<<prepBlocks, 256, 0, stream>>>(w0, w1, w2, w3, gw0, gw1, gw2, frag);

  int B = in_sizes[0] / 32;           // number of rows
  int blocks = B / TB;                // 2048 for B=131072
  moe_main<<<blocks, 256, 0, stream>>>(z, c, b0, b1, b2, b3,
                                       gb0, gb1, gb2, frag, (float*)d_out);
}
